// GroupedQueryAttention_3728031613198
// MI455X (gfx1250) — matrix operand order repaired and checked by reading, not compiled, not run on hardware
//
#include <hip/hip_runtime.h>
#include <hip/hip_bf16.h>
#include <stdint.h>
#include <stddef.h>

// ---------------- problem constants ----------------
#define D_MODEL   2048
#define T_SEQ     2048
#define NUM_HEADS 32
#define NUM_KV    8
#define HEAD_DIM  64
#define GQA_GROUP (NUM_HEADS / NUM_KV)
#define N_QKV     (NUM_HEADS * HEAD_DIM + 2 * NUM_KV * HEAD_DIM) // 3072
#define RMS_EPS   1e-6f

// ---------------- types for WMMA ----------------
typedef __bf16 bf16;
typedef bf16  bf16x16 __attribute__((ext_vector_type(16)));
typedef float f32x8   __attribute__((ext_vector_type(8)));
typedef unsigned int u32;
typedef u32 u32x4 __attribute__((ext_vector_type(4)));
typedef int  i32x4 __attribute__((ext_vector_type(4)));

union FragU { u32x4 q[2]; bf16x16 v; };

static __device__ __forceinline__ bf16 to_bf16(float f) { return (bf16)f; }

static __device__ __forceinline__ f32x8 wmma_bf16(const FragU& a, const FragU& b, f32x8 c) {
    // D = A(16x32 bf16) * B(32x16 bf16) + C(16x16 f32)
    return __builtin_amdgcn_wmma_f32_16x16x32_bf16(
        /*neg_a=*/false, a.v, /*neg_b=*/false, b.v,
        /*c_mod=*/(short)0, c, /*reuse_a=*/false, /*reuse_b=*/false);
}

// ---------------- CDNA5 async global->LDS copy (guarded) ----------------
#if defined(__gfx1250__) && __has_builtin(__builtin_amdgcn_global_load_async_to_lds_b128)
#define HAS_ASYNC_LDS 1
#else
#define HAS_ASYNC_LDS 0
#endif

#if HAS_ASYNC_LDS
typedef __attribute__((address_space(1))) i32x4 gi32x4;   // global int4
typedef __attribute__((address_space(3))) i32x4 li32x4;   // LDS int4
#endif

static __device__ __forceinline__ void cp16_g2l(const bf16* g, bf16* l) {
#if HAS_ASYNC_LDS
    __builtin_amdgcn_global_load_async_to_lds_b128(
        (gi32x4*)(const void*)g, (li32x4*)(void*)l, 0, 0);
#else
    *(u32x4*)l = *(const u32x4*)g;
#endif
}

template <int N>
static __device__ __forceinline__ void wait_async() {
#if HAS_ASYNC_LDS
#if __has_builtin(__builtin_amdgcn_s_wait_asynccnt)
    __builtin_amdgcn_s_wait_asynccnt((short)N);
#else
    asm volatile("s_wait_asynccnt %0" ::"i"(N) : "memory");
#endif
#endif
}

// ---------------- conversion / layout kernels ----------------
__global__ void f32_to_bf16_kernel(const float* __restrict__ in, bf16* __restrict__ out, int n) {
    for (int i = blockIdx.x * blockDim.x + threadIdx.x; i < n; i += gridDim.x * blockDim.x)
        out[i] = to_bf16(in[i]);
}

// WT[n][k] = W[k][n] for the concatenated [Wq | Wk | Wv] (columns 0..3071), K = 2048
__global__ void build_wqkvT_kernel(const float* __restrict__ Wq, const float* __restrict__ Wk,
                                   const float* __restrict__ Wv, bf16* __restrict__ WT) {
    int idx = blockIdx.x * blockDim.x + threadIdx.x; // over 3072*2048
    int n = idx >> 11;          // output row (original column), 0..3071
    int k = idx & 2047;         // original row
    float v;
    if (n < NUM_HEADS * HEAD_DIM)
        v = Wq[(size_t)k * (NUM_HEADS * HEAD_DIM) + n];
    else if (n < NUM_HEADS * HEAD_DIM + NUM_KV * HEAD_DIM)
        v = Wk[(size_t)k * (NUM_KV * HEAD_DIM) + (n - NUM_HEADS * HEAD_DIM)];
    else
        v = Wv[(size_t)k * (NUM_KV * HEAD_DIM) + (n - NUM_HEADS * HEAD_DIM - NUM_KV * HEAD_DIM)];
    WT[(size_t)idx] = to_bf16(v);
}

__global__ void build_woT_kernel(const float* __restrict__ Wo, bf16* __restrict__ WT) {
    int idx = blockIdx.x * blockDim.x + threadIdx.x; // over 2048*2048
    int n = idx >> 11;
    int k = idx & 2047;
    WT[(size_t)idx] = to_bf16(Wo[(size_t)k * D_MODEL + n]);
}

// ---------------- generic bf16 WMMA GEMM ----------------
// C[M,N](f32) = A[M,K](bf16 row-major) * BT[N,K]^T (bf16 pre-transposed row-major)
// block = 256 threads = 8 waves (2 x 4); wave tile = 64(M) x 64(N); block tile = 128 x 256.
// Double-buffered register fragments: loads for step k+32 are issued before WMMAs of step k.
__global__ void __launch_bounds__(256)
gemm_bf16_wmma(const bf16* __restrict__ A, const bf16* __restrict__ BT,
               float* __restrict__ C, int M, int N, int K) {
    const int lane = threadIdx.x & 31;
    const int wave = threadIdx.x >> 5;
    const int wm   = wave >> 2;            // 0..1
    const int wn   = wave & 3;             // 0..3
    const int m0   = blockIdx.x * 128 + wm * 64;
    const int n0   = blockIdx.y * 256 + wn * 64;
    const int half = lane >> 4;            // K sub-phase
    const int r    = lane & 15;            // row / column within 16-tile

    f32x8 acc[4][4] = {};                  // [mi][nf] 16x16 tiles

    const bf16* aPtr[4];
    const bf16* bPtr[4];
#pragma unroll
    for (int i = 0; i < 4; ++i)
        aPtr[i] = A + (size_t)(m0 + i * 16 + r) * K + half * 16;
#pragma unroll
    for (int f = 0; f < 4; ++f)
        bPtr[f] = BT + (size_t)(n0 + f * 16 + r) * K + half * 16;

    auto load_frags = [&](FragU (&a)[4], FragU (&b)[4], int k0) {
#pragma unroll
        for (int i = 0; i < 4; ++i) {
            a[i].q[0] = *(const u32x4*)(aPtr[i] + k0);        // K {0..7} / {16..23}
            a[i].q[1] = *(const u32x4*)(aPtr[i] + k0 + 8);    // K {8..15} / {24..31}
        }
#pragma unroll
        for (int f = 0; f < 4; ++f) {
            b[f].q[0] = *(const u32x4*)(bPtr[f] + k0);        // K {0..7} / {16..23}
            b[f].q[1] = *(const u32x4*)(bPtr[f] + k0 + 8);    // K {8..15} / {24..31}
        }
    };
    auto do_wmma = [&](FragU (&a)[4], FragU (&b)[4]) {
#pragma unroll
        for (int i = 0; i < 4; ++i)
#pragma unroll
            for (int f = 0; f < 4; ++f)
                acc[i][f] = wmma_bf16(a[i], b[f], acc[i][f]);
    };

    FragU a0[4], b0[4], a1[4], b1[4];
    load_frags(a0, b0, 0);
    for (int k0 = 0; k0 < K; k0 += 64) {
        __builtin_prefetch(aPtr[0] + k0 + 1024, 0, 1);
        __builtin_prefetch(bPtr[0] + k0 + 1024, 0, 1);
        load_frags(a1, b1, k0 + 32);       // always in range: K % 64 == 0
        do_wmma(a0, b0);
        if (k0 + 64 < K) load_frags(a0, b0, k0 + 64);
        do_wmma(a1, b1);
    }

    // C layout: lane r = column, VGPR rr = row (half selects rows 0-7 / 8-15)
#pragma unroll
    for (int i = 0; i < 4; ++i)
#pragma unroll
        for (int f = 0; f < 4; ++f) {
            float* cp = C + (size_t)(m0 + i * 16 + half * 8) * N + (n0 + f * 16 + r);
#pragma unroll
            for (int rr = 0; rr < 8; ++rr)
                cp[(size_t)rr * N] = acc[i][f][rr];
        }
}

// ---------------- RMSNorm + RoPE + layout for attention ----------------
// One wave per (token, unit): units 0..31 = Q heads, 32..39 = K heads, 40..47 = V heads.
// Lane L handles dims L and L+32 (the RoPE rotation pair).
__global__ void __launch_bounds__(256)
rms_rope_kernel(const float* __restrict__ qkv,
                const float* __restrict__ cosT, const float* __restrict__ sinT,
                const float* __restrict__ q_scale, const float* __restrict__ k_scale,
                bf16* __restrict__ Qb, bf16* __restrict__ Kb, bf16* __restrict__ Vt) {
    const int lane = threadIdx.x & 31;
    const int wave = threadIdx.x >> 5;
    const int gw   = blockIdx.x * 8 + wave;
    const int t    = gw / 48;
    const int unit = gw % 48;
    const float* row = qkv + (size_t)t * N_QKV;

    if (unit < NUM_HEADS + NUM_KV) {
        const bool isQ = unit < NUM_HEADS;
        const int h = isQ ? unit : unit - NUM_HEADS;
        const int base = isQ ? h * HEAD_DIM : NUM_HEADS * HEAD_DIM + h * HEAD_DIM;
        const float* sc = isQ ? q_scale : k_scale;
        float x1 = row[base + lane];
        float x2 = row[base + lane + 32];
        float ss = x1 * x1 + x2 * x2;
#pragma unroll
        for (int off = 1; off < 32; off <<= 1) ss += __shfl_xor(ss, off, 32);
        float rms = rsqrtf(ss * (1.0f / HEAD_DIM) + RMS_EPS);
        float y1 = x1 * rms * sc[lane];
        float y2 = x2 * rms * sc[lane + 32];
        float c1 = cosT[t * HEAD_DIM + lane],      s1 = sinT[t * HEAD_DIM + lane];
        float c2 = cosT[t * HEAD_DIM + lane + 32], s2 = sinT[t * HEAD_DIM + lane + 32];
        float o1 = y1 * c1 - y2 * s1;   // rotate_half: first half pairs with -x2
        float o2 = y2 * c2 + y1 * s2;   // second half pairs with +x1
        if (isQ) {                       // fold softmax 1/sqrt(HEAD_DIM) into Q
            o1 *= 0.125f; o2 *= 0.125f;
            bf16* q = Qb + ((size_t)h * T_SEQ + t) * HEAD_DIM;
            q[lane] = to_bf16(o1); q[lane + 32] = to_bf16(o2);
        } else {
            bf16* k = Kb + ((size_t)h * T_SEQ + t) * HEAD_DIM;
            k[lane] = to_bf16(o1); k[lane + 32] = to_bf16(o2);
        }
    } else {
        const int h = unit - NUM_HEADS - NUM_KV;
        const int base = NUM_HEADS * HEAD_DIM + NUM_KV * HEAD_DIM + h * HEAD_DIM;
        float x1 = row[base + lane];
        float x2 = row[base + lane + 32];
        bf16* v = Vt + (size_t)h * HEAD_DIM * T_SEQ;   // V transposed: [kv][d][T]
        v[(size_t)lane * T_SEQ + t]        = to_bf16(x1);
        v[(size_t)(lane + 32) * T_SEQ + t] = to_bf16(x2);
    }
}

// ---------------- flash attention (causal, GQA), block-cooperative ----------------
// One block = one head x 8 consecutive 16-row query tiles (one tile per wave).
// All 8 waves share the same KV head, so K (32x64) and V^T (64x32) tiles are staged
// into LDS once per key block via async global->LDS copies (ping-pong double buffer).
__global__ void __launch_bounds__(256)
attn_kernel(const bf16* __restrict__ Qb, const bf16* __restrict__ Kb,
            const bf16* __restrict__ Vt, bf16* __restrict__ ctxb) {
    __shared__ __attribute__((aligned(16))) bf16 Klds[2][32 * 64];   // [buf][kt][d]
    __shared__ __attribute__((aligned(16))) bf16 Vlds[2][64 * 32];   // [buf][d][kt]
    __shared__ __attribute__((aligned(16))) bf16 pbuf[8][16][32];    // per-wave P transpose

    const int lane = threadIdx.x & 31;
    const int wave = threadIdx.x >> 5;
    const int h    = blockIdx.x >> 4;              // head 0..31
    const int qg   = blockIdx.x & 15;              // q-tile group
    const int qt   = qg * 8 + wave;                // this wave's 16-row query tile
    const int kv   = h / GQA_GROUP;
    const int half = lane >> 4;
    const int r    = lane & 15;
    const int q_row_base = qt * 16 + half * 8;

    // Q fragments for d = 0..31 and 32..63 (A layout)
    const bf16* qRow = Qb + ((size_t)h * T_SEQ + qt * 16 + r) * HEAD_DIM + half * 16;
    FragU qa[2];
    qa[0].q[0] = *(const u32x4*)(qRow);
    qa[0].q[1] = *(const u32x4*)(qRow + 8);
    qa[1].q[0] = *(const u32x4*)(qRow + 32);
    qa[1].q[1] = *(const u32x4*)(qRow + 40);

    f32x8 acc[4] = {};                             // ctx tile: d columns f*16 + r
    float mrow[8], lrow[8];
#pragma unroll
    for (int i = 0; i < 8; ++i) { mrow[i] = -1e30f; lrow[i] = 0.0f; }

    const int nkb_wave = (qt >> 1) + 1;            // key blocks this wave needs (causal)
    const int nkb_max  = qg * 4 + 4;               // key blocks the whole block stages

    // stage key block kb into LDS buffer buf: 256 threads x 16B for K, same for V
    auto stage = [&](int kb, int buf) {
        const int c = threadIdx.x;                 // chunk 0..255
        const bf16* ksrc = Kb + ((size_t)kv * T_SEQ + kb * 32) * HEAD_DIM + c * 8;
        cp16_g2l(ksrc, &Klds[buf][c * 8]);
        const int d = c >> 2, part = c & 3;
        const bf16* vsrc = Vt + ((size_t)kv * HEAD_DIM + d) * T_SEQ + kb * 32 + part * 8;
        cp16_g2l(vsrc, &Vlds[buf][d * 32 + part * 8]);
    };

    stage(0, 0);
    for (int kb = 0; kb < nkb_max; ++kb) {
        const int cur = kb & 1;
        if (kb + 1 < nkb_max) { stage(kb + 1, cur ^ 1); wait_async<2>(); }
        else                  { wait_async<0>(); }
        __syncthreads();

        if (kb < nkb_wave) {
            const bf16* Kt    = Klds[cur];
            const bf16* Vtile = Vlds[cur];
            const int kt0 = kb * 32;

            // ---- scores: S(16q x 32k) as two 16x16 tiles; batch all K frag loads ----
            FragU bk[2][2];
#pragma unroll
            for (int c = 0; c < 2; ++c) {
                const bf16* kr = Kt + (c * 16 + r) * 64 + half * 16;
                bk[c][0].q[0] = *(const u32x4*)(kr);        // d {0..15} / {16..31}
                bk[c][0].q[1] = *(const u32x4*)(kr + 8);
                bk[c][1].q[0] = *(const u32x4*)(kr + 32);   // d {32..47} / {48..63}
                bk[c][1].q[1] = *(const u32x4*)(kr + 40);
            }
            f32x8 s[2] = {};
#pragma unroll
            for (int c = 0; c < 2; ++c) {
                s[c] = wmma_bf16(qa[0], bk[c][0], s[c]);
                s[c] = wmma_bf16(qa[1], bk[c][1], s[c]);
            }
            // ---- load V frags early so DS latency hides under softmax VALU ----
            FragU bv[4];
#pragma unroll
            for (int f = 0; f < 4; ++f) {
                const bf16* vp = Vtile + (f * 16 + r) * 32 + half * 16;
                bv[f].q[0] = *(const u32x4*)(vp);
                bv[f].q[1] = *(const u32x4*)(vp + 8);
            }
            // ---- causal mask + online softmax ----
            const int colt = kt0 + r;
#pragma unroll
            for (int rr = 0; rr < 8; ++rr) {
                const int qrow = q_row_base + rr;
                float s0 = (colt      <= qrow) ? s[0][rr] : -1e30f;
                float s1 = (colt + 16 <= qrow) ? s[1][rr] : -1e30f;
                float mx = fmaxf(s0, s1);
#pragma unroll
                for (int off = 1; off < 16; off <<= 1) mx = fmaxf(mx, __shfl_xor(mx, off, 32));
                float mnew = fmaxf(mrow[rr], mx);
                float p0   = __expf(s0 - mnew);
                float p1   = __expf(s1 - mnew);
                float psum = p0 + p1;
#pragma unroll
                for (int off = 1; off < 16; off <<= 1) psum += __shfl_xor(psum, off, 32);
                float alpha = __expf(mrow[rr] - mnew);
                mrow[rr] = mnew;
                lrow[rr] = lrow[rr] * alpha + psum;
#pragma unroll
                for (int f = 0; f < 4; ++f) acc[f][rr] *= alpha;
                const int prow = half * 8 + rr;          // P: C layout -> LDS
                pbuf[wave][prow][r]      = to_bf16(p0);
                pbuf[wave][prow][r + 16] = to_bf16(p1);
            }
            // ---- P(16x32) * V(32x64): P back in A layout from LDS ----
            FragU pa;
            {
                const bf16* pr = &pbuf[wave][r][half * 16];
                pa.q[0] = *(const u32x4*)(pr);        // k {0..7} / {16..23}
                pa.q[1] = *(const u32x4*)(pr + 8);    // k {8..15} / {24..31}
            }
#pragma unroll
            for (int f = 0; f < 4; ++f)
                acc[f] = wmma_bf16(pa, bv[f], acc[f]);
        }
        __syncthreads();
    }

    // ---- normalize and store ctx (bf16, [T][D_MODEL]) ----
#pragma unroll
    for (int rr = 0; rr < 8; ++rr) {
        float inv = 1.0f / lrow[rr];
        const size_t rowoff = (size_t)(q_row_base + rr) * D_MODEL + h * HEAD_DIM + r;
#pragma unroll
        for (int f = 0; f < 4; ++f)
            ctxb[rowoff + f * 16] = to_bf16(acc[f][rr] * inv);
    }
}

// ---------------- host launcher ----------------
extern "C" void kernel_launch(void* const* d_in, const int* in_sizes, int n_in,
                              void* d_out, int out_size, void* d_ws, size_t ws_size,
                              hipStream_t stream) {
    const float* x    = (const float*)d_in[0];
    // d_in[1] = mask (bool) — causality implemented directly, unused
    const float* cosT = (const float*)d_in[2];
    const float* sinT = (const float*)d_in[3];
    const float* Wq   = (const float*)d_in[4];
    const float* Wk   = (const float*)d_in[5];
    const float* Wv   = (const float*)d_in[6];
    const float* Wo   = (const float*)d_in[7];
    const float* qs   = (const float*)d_in[8];
    const float* ks   = (const float*)d_in[9];

    char* ws = (char*)d_ws;
    size_t off = 0;
    auto alloc = [&](size_t bytes) -> void* {
        void* p = ws + off;
        off = (off + bytes + 255) & ~(size_t)255;
        return p;
    };
    bf16*  Xb     = (bf16*) alloc((size_t)T_SEQ * D_MODEL * sizeof(bf16));
    bf16*  WqkvT  = (bf16*) alloc((size_t)N_QKV * D_MODEL * sizeof(bf16));
    bf16*  WoT    = (bf16*) alloc((size_t)D_MODEL * D_MODEL * sizeof(bf16));
    float* qkv    = (float*)alloc((size_t)T_SEQ * N_QKV * sizeof(float));
    bf16*  Qb     = (bf16*) alloc((size_t)NUM_HEADS * T_SEQ * HEAD_DIM * sizeof(bf16));
    bf16*  Kb     = (bf16*) alloc((size_t)NUM_KV * T_SEQ * HEAD_DIM * sizeof(bf16));
    bf16*  Vt     = (bf16*) alloc((size_t)NUM_KV * HEAD_DIM * T_SEQ * sizeof(bf16));
    bf16*  ctxb   = (bf16*) alloc((size_t)T_SEQ * D_MODEL * sizeof(bf16));
    (void)ws_size; (void)in_sizes; (void)n_in; (void)out_size;

    // 1) activations -> bf16
    f32_to_bf16_kernel<<<4096, 256, 0, stream>>>(x, Xb, T_SEQ * D_MODEL);
    // 2) weights -> transposed bf16
    build_wqkvT_kernel<<<(N_QKV * D_MODEL) / 256, 256, 0, stream>>>(Wq, Wk, Wv, WqkvT);
    build_woT_kernel<<<(D_MODEL * D_MODEL) / 256, 256, 0, stream>>>(Wo, WoT);
    // 3) fused QKV projection: [2048,2048] x [2048,3072]
    gemm_bf16_wmma<<<dim3(T_SEQ / 128, N_QKV / 256), 256, 0, stream>>>(
        Xb, WqkvT, qkv, T_SEQ, N_QKV, D_MODEL);
    // 4) RMSNorm + RoPE + attention layouts
    rms_rope_kernel<<<(T_SEQ * 48) / 8, 256, 0, stream>>>(qkv, cosT, sinT, qs, ks, Qb, Kb, Vt);
    // 5) causal GQA flash attention (block-cooperative, async K/V staging)
    attn_kernel<<<NUM_HEADS * (T_SEQ / 16 / 8), 256, 0, stream>>>(Qb, Kb, Vt, ctxb);
    // 6) output projection: [2048,2048] x [2048,2048] -> f32 d_out
    gemm_bf16_wmma<<<dim3(T_SEQ / 128, D_MODEL / 256), 256, 0, stream>>>(
        ctxb, WoT, (float*)d_out, T_SEQ, D_MODEL, D_MODEL);
}
